// to_vec_21603685499545
// MI455X (gfx1250) — matrix operand order repaired and checked by reading, not compiled, not run on hardware
//
#include <hip/hip_runtime.h>

typedef __attribute__((ext_vector_type(16))) _Float16 v16h;
typedef __attribute__((ext_vector_type(8)))  _Float16 v8h;
typedef __attribute__((ext_vector_type(4)))  _Float16 v4h;
typedef __attribute__((ext_vector_type(8)))  float    v8f;
typedef __attribute__((ext_vector_type(4)))  float    v4f;

#define Bdim 8
#define Gdim 16
#define Ndim 1024
#define Cdim 256
#define TILE_M 64
#define XSTR 264            // 256 + 8 f16 pad -> row stride 132 dwords, conflict-free fragment loads
#define WSTR 264
#define NROWTILES (Ndim / TILE_M)   // 16

__global__ __launch_bounds__(256) void vn_atten_kernel(
    const float* __restrict__ x, const float* __restrict__ vs,
    const float* __restrict__ Wfc, const float* __restrict__ bfc,
    const float* __restrict__ Wvn, float* __restrict__ out)
{
    // ~240 KB static LDS: one workgroup per WGP (320 KB available on gfx1250).
    __shared__ _Float16 sW[Cdim * WSTR];            // W_fc as f16, row-major [o][k], padded
    __shared__ _Float16 sX[2][TILE_M * XSTR];       // double-buffered x tile (f16, padded)
    __shared__ float sB[Cdim];
    __shared__ float sV[Ndim * 3];
    __shared__ float sMaxPart[4][Cdim];
    __shared__ float sMax[Cdim];
    __shared__ float sSumPart[4][Cdim];
    __shared__ float sAPart[4][Cdim][3];
    __shared__ float sOut[Cdim][3];

    const int tid  = threadIdx.x;
    const int lane = tid & 31;
    const int wave = tid >> 5;
    const int rs   = wave >> 1;   // row-subtile 0..3 (16 rows each)
    const int ch   = wave & 1;    // column half 0..1 (128 cols each)
    const int l15  = lane & 15;
    const int hi   = lane >> 4;   // 0 | 1 (upper half-wave)
    const int bg   = blockIdx.x;  // 0..127

    const float* xg = x  + (size_t)bg * Ndim * Cdim;
    const float* vg = vs + (size_t)bg * Ndim * 3;

    // ---- one-time LDS init: W_fc -> f16, bias, v, max partials ----
    {
        const v4f* wsrc = (const v4f*)Wfc;
        #pragma unroll
        for (int i = 0; i < 64; ++i) {
            int idx = tid + i * 256;            // over 16384 vec4
            int row = idx >> 6, c4 = idx & 63;
            v4f w = wsrc[idx];
            v4h h = { (_Float16)w[0], (_Float16)w[1], (_Float16)w[2], (_Float16)w[3] };
            *(v4h*)(sW + row * WSTR + (c4 << 2)) = h;
        }
        #pragma unroll
        for (int i = 0; i < 12; ++i) sV[tid + i * 256] = vg[tid + i * 256];
        sB[tid] = bfc[tid];
        #pragma unroll
        for (int r = 0; r < 4; ++r) sMaxPart[r][tid] = -1e30f;
    }

    // staging registers for the next x tile (f32 loaded, converted to f16)
    v4h xreg[16];

    auto load_regs = [&](int tb) {
        const v4f* src = (const v4f*)xg + tb * (TILE_M * (Cdim / 4));
        #pragma unroll
        for (int i = 0; i < 16; ++i) {
            v4f t = src[tid + i * 256];
            xreg[i] = { (_Float16)t[0], (_Float16)t[1], (_Float16)t[2], (_Float16)t[3] };
        }
    };
    auto store_lds = [&](int buf) {
        #pragma unroll
        for (int i = 0; i < 16; ++i) {
            int idx = tid + i * 256;
            int row = idx >> 6, c4 = idx & 63;
            *(v4h*)(sX[buf] + row * XSTR + (c4 << 2)) = xreg[i];
        }
    };

    // A fragment (16x32 f16): lane<16 holds K 0..7,16..23; lane>=16 K 8..15,24..31
    auto ld_a = [&](const _Float16* xb, int kk) -> v16h {
        const _Float16* ap = xb + (rs * 16 + l15) * XSTR + kk * 32 + (hi << 3);
        v8h lo = *(const v8h*)ap;
        v8h hh = *(const v8h*)(ap + 16);
        return __builtin_shufflevector(lo, hh,
            0,1,2,3,4,5,6,7,8,9,10,11,12,13,14,15);
    };
    // B fragment (32x16): col = W_fc row c; lane<16 K 0..7,16..23; lane>=16 K 8..15,24..31
    auto ld_b = [&](int kk, int t) -> v16h {
        const _Float16* bp = sW + (ch * 128 + t * 16 + l15) * WSTR + kk * 32 + (hi << 3);
        v8h lo = *(const v8h*)bp;
        v8h hh = *(const v8h*)(bp + 16);
        return __builtin_shufflevector(lo, hh,
            0,1,2,3,4,5,6,7,8,9,10,11,12,13,14,15);
    };

    // q-tile GEMM: 16 rows x 128 cols per wave, K=256 in 8 steps of 32.
    // Explicit depth-2 software pipeline on B fragments (and depth-1 on A) so every
    // WMMA only waits on LDS loads issued >=2 fragments earlier (s_wait_dscnt >= 2).
    auto mm_tile = [&](v8f* acc, const _Float16* xb) {
        v16h a_cur = ld_a(xb, 0);
        v16h b0 = ld_b(0, 0);
        v16h b1 = ld_b(0, 1);
        #pragma unroll
        for (int kk = 0; kk < 8; ++kk) {
            v16h a_nxt = a_cur;
            if (kk < 7) a_nxt = ld_a(xb, kk + 1);
            #pragma unroll
            for (int t = 0; t < 8; ++t) {
                v16h b_new = b1;
                const int ni = kk * 8 + t + 2;          // fragment index 2 ahead
                if (ni < 64) b_new = ld_b(ni >> 3, ni & 7);
                acc[t] = __builtin_amdgcn_wmma_f32_16x16x32_f16(
                    false, a_cur, false, b0, (short)0, acc[t], false, false);
                b0 = b1; b1 = b_new;
            }
            a_cur = a_nxt;
        }
    };

    // ================= PASS 1: column maxes of q =================
    load_regs(0);
    store_lds(0);
    #pragma unroll 1
    for (int tb = 0; tb < NROWTILES; ++tb) {
        __syncthreads();                       // buffer (tb&1) now visible; prior readers done
        const int cur = tb & 1;
        if (tb + 1 < NROWTILES) load_regs(tb + 1);   // global loads overlap WMMA below
        v8f acc[8] = {};
        mm_tile(acc, sX[cur]);
        #pragma unroll
        for (int t = 0; t < 8; ++t) {
            const int c = ch * 128 + t * 16 + l15;
            const float bias = sB[c];
            float m = -1e30f;
            #pragma unroll
            for (int j = 0; j < 8; ++j) m = fmaxf(m, acc[t][j] + bias);
            m = fmaxf(m, __shfl_xor(m, 16, 32));     // join the two 8-row halves
            if (lane < 16) sMaxPart[rs][c] = fmaxf(sMaxPart[rs][c], m);
        }
        if (tb + 1 < NROWTILES) store_lds((tb + 1) & 1);
    }
    __syncthreads();
    {
        sMax[tid] = fmaxf(fmaxf(sMaxPart[0][tid], sMaxPart[1][tid]),
                          fmaxf(sMaxPart[2][tid], sMaxPart[3][tid]));
        #pragma unroll
        for (int r = 0; r < 4; ++r) {
            sSumPart[r][tid] = 0.0f;
            sAPart[r][tid][0] = 0.0f; sAPart[r][tid][1] = 0.0f; sAPart[r][tid][2] = 0.0f;
        }
    }
    __syncthreads();

    // ==== PASS 2: p = exp(q-max); accumulate sum_n p and sum_n p*x*v ====
    load_regs(0);
    store_lds(0);
    #pragma unroll 1
    for (int tb = 0; tb < NROWTILES; ++tb) {
        __syncthreads();
        const int cur = tb & 1;
        if (tb + 1 < NROWTILES) load_regs(tb + 1);
        v8f acc[8] = {};
        mm_tile(acc, sX[cur]);
        #pragma unroll
        for (int t = 0; t < 8; ++t) {
            const int c = ch * 128 + t * 16 + l15;
            const float bias = sB[c];
            const float mx = sMax[c];
            float s = 0.f, a0 = 0.f, a1 = 0.f, a2 = 0.f;
            #pragma unroll
            for (int j = 0; j < 8; ++j) {
                const int mrow = rs * 16 + (hi ? 8 + j : j);
                const float p  = __expf(acc[t][j] + bias - mx);
                const float xv = (float)sX[cur][mrow * XSTR + c];
                const float pw = p * xv;
                const int n = tb * TILE_M + mrow;
                a0 += pw * sV[n * 3 + 0];
                a1 += pw * sV[n * 3 + 1];
                a2 += pw * sV[n * 3 + 2];
                s  += p;
            }
            s  += __shfl_xor(s,  16, 32);
            a0 += __shfl_xor(a0, 16, 32);
            a1 += __shfl_xor(a1, 16, 32);
            a2 += __shfl_xor(a2, 16, 32);
            if (lane < 16) {
                sSumPart[rs][c]  += s;
                sAPart[rs][c][0] += a0;
                sAPart[rs][c][1] += a1;
                sAPart[rs][c][2] += a2;
            }
        }
        if (tb + 1 < NROWTILES) store_lds((tb + 1) & 1);
    }
    __syncthreads();

    // ---- normalize: out[c,d] = A[c,d] / sum[c] ----
    {
        float s = sSumPart[0][tid] + sSumPart[1][tid] + sSumPart[2][tid] + sSumPart[3][tid];
        float inv = 1.0f / s;
        #pragma unroll
        for (int d = 0; d < 3; ++d)
            sOut[tid][d] = (sAPart[0][tid][d] + sAPart[1][tid][d] +
                            sAPart[2][tid][d] + sAPart[3][tid][d]) * inv;
    }
    __syncthreads();

    // ---- W_vn: thread tid = output channel o; out2[o,d] = sum_c Wvn[o,c]*sOut[c,d] ----
    {
        const v4f* wrow = (const v4f*)(Wvn + (size_t)tid * Cdim);
        float r0 = 0.f, r1 = 0.f, r2 = 0.f;
        #pragma unroll 4
        for (int c4 = 0; c4 < 64; ++c4) {
            v4f w = wrow[c4];
            #pragma unroll
            for (int j = 0; j < 4; ++j) {
                const int c = (c4 << 2) + j;
                const float ww = w[j];
                r0 += ww * sOut[c][0];
                r1 += ww * sOut[c][1];
                r2 += ww * sOut[c][2];
            }
        }
        float* og = out + ((size_t)bg * Cdim + tid) * 3;
        og[0] = r0; og[1] = r1; og[2] = r2;
    }
}

extern "C" void kernel_launch(void* const* d_in, const int* in_sizes, int n_in,
                              void* d_out, int out_size, void* d_ws, size_t ws_size,
                              hipStream_t stream) {
    const float* x   = (const float*)d_in[0];
    const float* vs  = (const float*)d_in[1];
    const float* Wfc = (const float*)d_in[2];
    const float* bfc = (const float*)d_in[3];
    const float* Wvn = (const float*)d_in[4];
    float* out = (float*)d_out;
    vn_atten_kernel<<<dim3(Bdim * Gdim), dim3(256), 0, stream>>>(x, vs, Wfc, bfc, Wvn, out);
}
